// GNNViewCompletionModule_22677427323619
// MI455X (gfx1250) — matrix operand order repaired and checked by reading, not compiled, not run on hardware
//
#include <hip/hip_runtime.h>
#include <hip/hip_bf16.h>
#include <cstdint>

// ---------------------------------------------------------------------------
// Types for CDNA5 WMMA (gfx1250, wave32)
// ---------------------------------------------------------------------------
typedef __attribute__((ext_vector_type(16))) __bf16          v16bf;
typedef __attribute__((ext_vector_type(16))) unsigned short  v16u;
typedef __attribute__((ext_vector_type(8)))  unsigned short  v8u;
typedef __attribute__((ext_vector_type(8)))  float           v8f;

#define B_DIM   16384
#define V_DIM   4
#define NNODES  (B_DIM * V_DIM)   // 65536
#define DM      128               // D_MODEL
#define HIDF    64                // HID
#define NHEADS  8
#define OUT_D   128

// ---------------------------------------------------------------------------
// bf16 helpers (memory type = unsigned short so host pass stays trivial)
// ---------------------------------------------------------------------------
__device__ __forceinline__ float bf2f(unsigned short u) {
    return __uint_as_float(((unsigned)u) << 16);
}
__device__ __forceinline__ unsigned short f2bf(float f) {
    unsigned x = __float_as_uint(f);
    // round-to-nearest-even
    unsigned r = (x + 0x7fffu + ((x >> 16) & 1u)) >> 16;
    return (unsigned short)r;
}

__device__ __forceinline__ void storeC(float* p, float v)          { *p = v; }
__device__ __forceinline__ void storeC(unsigned short* p, float v) { *p = f2bf(v); }

// ---------------------------------------------------------------------------
// A-fragment loaders. ISA 16-bit A 16x32 layout (per lane, hi = lane>=16):
//   elements j=0..7  -> K = kk + j      + 8*hi
//   elements j=8..15 -> K = kk + j + 8  + 8*hi   (i.e. base+16 .. base+23)
// So each lane reads two contiguous 8-element runs: [base, base+8) and
// [base+16, base+24) where base = kk + 8*hi.
// ---------------------------------------------------------------------------
__device__ __forceinline__ v16bf load_afrag(const float* p) {
    const float4* q0 = reinterpret_cast<const float4*>(p);
    const float4* q1 = reinterpret_cast<const float4*>(p + 16);
    float4 a0 = q0[0], a1 = q0[1];
    float4 a2 = q1[0], a3 = q1[1];
    v16u u;
    u[0]  = f2bf(a0.x); u[1]  = f2bf(a0.y); u[2]  = f2bf(a0.z); u[3]  = f2bf(a0.w);
    u[4]  = f2bf(a1.x); u[5]  = f2bf(a1.y); u[6]  = f2bf(a1.z); u[7]  = f2bf(a1.w);
    u[8]  = f2bf(a2.x); u[9]  = f2bf(a2.y); u[10] = f2bf(a2.z); u[11] = f2bf(a2.w);
    u[12] = f2bf(a3.x); u[13] = f2bf(a3.y); u[14] = f2bf(a3.z); u[15] = f2bf(a3.w);
    return __builtin_bit_cast(v16bf, u);
}
__device__ __forceinline__ v16bf load_afrag(const unsigned short* p) {
    v8u lo = *reinterpret_cast<const v8u*>(p);
    v8u hi = *reinterpret_cast<const v8u*>(p + 16);
    v16u u;
#pragma unroll
    for (int j = 0; j < 8; ++j) { u[j] = lo[j]; u[j + 8] = hi[j]; }
    return __builtin_bit_cast(v16bf, u);
}

// ---------------------------------------------------------------------------
// Pack f32 weight W[K,N] (row-major) into bf16 B-fragment order:
// P[((tn*(K/32)+kb)*32 + lane)*16 + j] = W[kb*32 + (j&7) + 8*(lane>=16) + 16*(j>=8)][tn*16 + lane%16]
// (ISA B layout: lane holds column n = lane%16; lanes>=16 hold K 8..15 and 24..31 of each block.)
// ---------------------------------------------------------------------------
__global__ void pack_weight_bf16(const float* __restrict__ W,
                                 unsigned short* __restrict__ P,
                                 int K, int N) {
    int idx = blockIdx.x * blockDim.x + threadIdx.x;
    if (idx >= K * N) return;
    const int j    = idx & 15;
    const int lane = (idx >> 4) & 31;
    const int rest = idx >> 9;
    const int kblocks = K >> 5;
    const int kb = rest % kblocks;
    const int tn = rest / kblocks;
    const int k  = (kb << 5) + (j & 7) + ((lane >> 4) << 3) + ((j >> 3) << 4);
    const int n  = (tn << 4) + (lane & 15);
    P[idx] = f2bf(W[(size_t)k * N + n]);
}

// ---------------------------------------------------------------------------
// Generic bf16 WMMA GEMM:  C[M, ldc] = A[M,K] @ Bpacked + bias
// 4 waves / block; each wave computes a 16 x (16*NTILES) strip using
// v_wmma_f32_16x16x32_bf16. M must be a multiple of 64, K of 32, N of 16.
// PERM: final layer writes row m -> (m % B_DIM)*V_DIM + m/B_DIM  ([B,V,OUT]).
// ---------------------------------------------------------------------------
template <typename TA, typename TC, int NTILES, bool PERM>
__global__ __launch_bounds__(128)
void gemm_bf16_wmma(const TA* __restrict__ A,
                    const unsigned short* __restrict__ Bp,
                    const float* __restrict__ bias,
                    TC* __restrict__ C,
                    int K, int lda, int ldc) {
    const int wave = threadIdx.x >> 5;
    const int lane = threadIdx.x & 31;
    const int hi   = lane >> 4;          // K-half selector
    const int col  = lane & 15;          // N (and A-row) index within tile
    const int m0   = blockIdx.x * 64 + wave * 16;
    const int tn0  = blockIdx.y * NTILES;
    const int kblocks = K >> 5;

    const TA* Arow = A + (size_t)(m0 + col) * lda;

    v8f acc[NTILES];
#pragma unroll
    for (int t = 0; t < NTILES; ++t) {
        v8f z = {0.f, 0.f, 0.f, 0.f, 0.f, 0.f, 0.f, 0.f};
        acc[t] = z;
    }

    for (int kb = 0; kb < kblocks; ++kb) {
        const int kk = (kb << 5) + (hi << 3);
        const v16bf a = load_afrag(Arow + kk);
#pragma unroll
        for (int t = 0; t < NTILES; ++t) {
            const v16u bu = *reinterpret_cast<const v16u*>(
                Bp + ((((size_t)(tn0 + t) * kblocks + kb) * 32 + lane) << 4));
            const v16bf b = __builtin_bit_cast(v16bf, bu);
            acc[t] = __builtin_amdgcn_wmma_f32_16x16x32_bf16(
                false, a, false, b, (short)0, acc[t], false, false);
        }
    }

#pragma unroll
    for (int t = 0; t < NTILES; ++t) {
        const int n = (tn0 + t) * 16 + col;
        const float bv = bias ? bias[n] : 0.f;
#pragma unroll
        for (int r = 0; r < 8; ++r) {
            const int mr = m0 + (hi << 3) + r;   // D layout: m = r + 8*hi
            float v = acc[t][r] + bv;
            size_t row = PERM ? ((size_t)(mr & (B_DIM - 1)) * V_DIM + (mr >> 14))
                              : (size_t)mr;
            storeC(C + row * (size_t)ldc + n, v);
        }
    }
}

// ---------------------------------------------------------------------------
// GAT attention, exploiting the fixed graph: node group {4g..4g+3} is a
// clique with self-loops, so each dst's softmax is over the 4 group nodes.
//
// Single-pass, wave-per-(group,head): each lane keeps features {lane, lane+32}
// of all 4 group nodes in registers; a_src/a_dst dot products use a wave32
// butterfly all-reduce (__shfl_xor -> ds permute path). h1 is read exactly
// once, fully coalesced.
// ---------------------------------------------------------------------------
__device__ __forceinline__ float wave_allreduce_sum(float v) {
#pragma unroll
    for (int m = 16; m >= 1; m >>= 1) v += __shfl_xor(v, m, 32);
    return v;
}

// Layer 1: 8 heads x 64 feats, bf16 in/out, epilogue bias+ReLU.
__global__ __launch_bounds__(256)
void gat_attention_h8(const unsigned short* __restrict__ H,
                      const float* __restrict__ asrc,
                      const float* __restrict__ adst,
                      const float* __restrict__ bias,
                      unsigned short* __restrict__ O) {
    const int wid  = blockIdx.x * (blockDim.x >> 5) + (threadIdx.x >> 5);
    const int lane = threadIdx.x & 31;
    if (wid >= B_DIM * NHEADS) return;          // grid sized so this never splits a wave
    const int g = wid >> 3;
    const int h = wid & 7;
    const int ld = NHEADS * HIDF;               // 512
    const size_t base = (size_t)g * 4 * ld + h * HIDF;

    float hv[4][2];
#pragma unroll
    for (int u = 0; u < 4; ++u) {
        const unsigned short* row = H + base + (size_t)u * ld;
        hv[u][0] = bf2f(row[lane]);
        hv[u][1] = bf2f(row[lane + 32]);
    }
    const float as0 = asrc[h * HIDF + lane], as1 = asrc[h * HIDF + lane + 32];
    const float ad0 = adst[h * HIDF + lane], ad1 = adst[h * HIDF + lane + 32];

    float s[4], d[4];
#pragma unroll
    for (int u = 0; u < 4; ++u) {
        s[u] = wave_allreduce_sum(hv[u][0] * as0 + hv[u][1] * as1);
        d[u] = wave_allreduce_sum(hv[u][0] * ad0 + hv[u][1] * ad1);
    }

    const float b0 = bias[h * HIDF + lane], b1 = bias[h * HIDF + lane + 32];
#pragma unroll
    for (int v = 0; v < 4; ++v) {
        float e[4], mx = -1e30f;
#pragma unroll
        for (int u = 0; u < 4; ++u) {
            float x = s[u] + d[v];
            e[u] = (x > 0.f) ? x : 0.2f * x;    // leaky_relu 0.2
            mx = fmaxf(mx, e[u]);
        }
        float den = 0.f;
#pragma unroll
        for (int u = 0; u < 4; ++u) { e[u] = __expf(e[u] - mx); den += e[u]; }
        const float inv = 1.f / (den + 1e-16f);
        float o0 = 0.f, o1 = 0.f;
#pragma unroll
        for (int u = 0; u < 4; ++u) { o0 += e[u] * hv[u][0]; o1 += e[u] * hv[u][1]; }
        o0 = fmaxf(o0 * inv + b0, 0.f);
        o1 = fmaxf(o1 * inv + b1, 0.f);
        unsigned short* orow = O + base + (size_t)v * ld;
        orow[lane]      = f2bf(o0);
        orow[lane + 32] = f2bf(o1);
    }
}

// Layer 2: 1 head x 64 feats, f32 in/out, wave-per-group.
__global__ __launch_bounds__(256)
void gat_attention_h1(const float* __restrict__ H,
                      const float* __restrict__ asrc,
                      const float* __restrict__ adst,
                      const float* __restrict__ bias,
                      float* __restrict__ O) {
    const int wid  = blockIdx.x * (blockDim.x >> 5) + (threadIdx.x >> 5);
    const int lane = threadIdx.x & 31;
    if (wid >= B_DIM) return;
    const size_t base = (size_t)wid * 4 * HIDF;

    float hv[4][2];
#pragma unroll
    for (int u = 0; u < 4; ++u) {
        const float* row = H + base + (size_t)u * HIDF;
        hv[u][0] = row[lane];
        hv[u][1] = row[lane + 32];
    }
    const float as0 = asrc[lane], as1 = asrc[lane + 32];
    const float ad0 = adst[lane], ad1 = adst[lane + 32];

    float s[4], d[4];
#pragma unroll
    for (int u = 0; u < 4; ++u) {
        s[u] = wave_allreduce_sum(hv[u][0] * as0 + hv[u][1] * as1);
        d[u] = wave_allreduce_sum(hv[u][0] * ad0 + hv[u][1] * ad1);
    }

    const float b0 = bias[lane], b1 = bias[lane + 32];
#pragma unroll
    for (int v = 0; v < 4; ++v) {
        float e[4], mx = -1e30f;
#pragma unroll
        for (int u = 0; u < 4; ++u) {
            float x = s[u] + d[v];
            e[u] = (x > 0.f) ? x : 0.2f * x;
            mx = fmaxf(mx, e[u]);
        }
        float den = 0.f;
#pragma unroll
        for (int u = 0; u < 4; ++u) { e[u] = __expf(e[u] - mx); den += e[u]; }
        const float inv = 1.f / (den + 1e-16f);
        float o0 = 0.f, o1 = 0.f;
#pragma unroll
        for (int u = 0; u < 4; ++u) { o0 += e[u] * hv[u][0]; o1 += e[u] * hv[u][1]; }
        float* orow = O + base + (size_t)v * HIDF;
        orow[lane]      = fmaxf(o0 * inv + b0, 0.f);
        orow[lane + 32] = fmaxf(o1 * inv + b1, 0.f);
    }
}

// ---------------------------------------------------------------------------
// Host-side pipeline
// ---------------------------------------------------------------------------
extern "C" void kernel_launch(void* const* d_in, const int* in_sizes, int n_in,
                              void* d_out, int out_size, void* d_ws, size_t ws_size,
                              hipStream_t stream) {
    (void)in_sizes; (void)n_in; (void)out_size; (void)ws_size;

    const int dims[4] = {256, 512, 384, 320};
    const float* x[4]  = {(const float*)d_in[0], (const float*)d_in[3],
                          (const float*)d_in[6], (const float*)d_in[9]};
    const float* W[4]  = {(const float*)d_in[1], (const float*)d_in[4],
                          (const float*)d_in[7], (const float*)d_in[10]};
    const float* bv[4] = {(const float*)d_in[2], (const float*)d_in[5],
                          (const float*)d_in[8], (const float*)d_in[11]};
    const float* g1W  = (const float*)d_in[12];
    const float* g1as = (const float*)d_in[13];
    const float* g1ad = (const float*)d_in[14];
    const float* g1b  = (const float*)d_in[15];
    const float* g2W  = (const float*)d_in[16];
    const float* g2as = (const float*)d_in[17];
    const float* g2ad = (const float*)d_in[18];
    const float* g2b  = (const float*)d_in[19];
    const float* fcW  = (const float*)d_in[20];
    const float* fcb  = (const float*)d_in[21];
    // d_in[22] = edge_index: structure is a fixed per-sample clique; not needed.

    char* ws = (char*)d_ws;
    size_t off = 0;
    auto alloc = [&](size_t bytes) -> void* {
        void* p = ws + off;
        off = (off + bytes + 255) & ~(size_t)255;
        return p;
    };
    unsigned short* nodes = (unsigned short*)alloc((size_t)NNODES * DM * 2);       // 16 MB bf16
    unsigned short* h1    = (unsigned short*)alloc((size_t)NNODES * 512 * 2);      // 64 MB bf16
    unsigned short* h1a   = (unsigned short*)alloc((size_t)NNODES * 512 * 2);      // 64 MB bf16
    float*          h2    = (float*)alloc((size_t)NNODES * HIDF * 4);              // 16 MB f32
    float*          h2a   = (float*)alloc((size_t)NNODES * HIDF * 4);              // 16 MB f32
    unsigned short* Wp[4];
    for (int v = 0; v < 4; ++v) Wp[v] = (unsigned short*)alloc((size_t)dims[v] * DM * 2);
    unsigned short* g1Wp = (unsigned short*)alloc((size_t)DM * 512 * 2);
    unsigned short* g2Wp = (unsigned short*)alloc((size_t)512 * HIDF * 2);
    unsigned short* fcWp = (unsigned short*)alloc((size_t)HIDF * OUT_D * 2);

    // --- pack all weights into bf16 B-fragment layout (tiny, L2 resident) ---
    for (int v = 0; v < 4; ++v) {
        int elems = dims[v] * DM;
        pack_weight_bf16<<<(elems + 255) / 256, 256, 0, stream>>>(W[v], Wp[v], dims[v], DM);
    }
    pack_weight_bf16<<<(DM * 512 + 255) / 256, 256, 0, stream>>>(g1W, g1Wp, DM, 512);
    pack_weight_bf16<<<(512 * HIDF + 255) / 256, 256, 0, stream>>>(g2W, g2Wp, 512, HIDF);
    pack_weight_bf16<<<(HIDF * OUT_D + 255) / 256, 256, 0, stream>>>(fcW, fcWp, HIDF, OUT_D);

    // --- per-view linears: nodes[v*B + i, :] = x_v @ W_v + b_v  (bf16 out) ---
    for (int v = 0; v < 4; ++v) {
        gemm_bf16_wmma<float, unsigned short, 8, false>
            <<<dim3(B_DIM / 64, 1), 128, 0, stream>>>(
                x[v], Wp[v], bv[v], nodes + (size_t)v * B_DIM * DM,
                dims[v], dims[v], DM);
    }

    // --- GAT1 linear: h1 = nodes @ gat1_W   [65536, 512] ---
    gemm_bf16_wmma<unsigned short, unsigned short, 8, false>
        <<<dim3(NNODES / 64, 512 / 128), 128, 0, stream>>>(
            nodes, g1Wp, nullptr, h1, DM, DM, 512);

    // --- GAT1 attention + bias + ReLU (wave per (group, head)) ---
    gat_attention_h8<<<(B_DIM * NHEADS) / 8, 256, 0, stream>>>(
        h1, g1as, g1ad, g1b, h1a);

    // --- GAT2 linear: h2 = h1a @ gat2_W   [65536, 64] ---
    gemm_bf16_wmma<unsigned short, float, 4, false>
        <<<dim3(NNODES / 64, 1), 128, 0, stream>>>(
            h1a, g2Wp, nullptr, h2, 512, 512, HIDF);

    // --- GAT2 attention + bias + ReLU (wave per group) ---
    gat_attention_h1<<<B_DIM / 8, 256, 0, stream>>>(
        h2, g2as, g2ad, g2b, h2a);

    // --- FC + output permutation to [B, V, OUT] ---
    gemm_bf16_wmma<float, float, 8, true>
        <<<dim3(NNODES / 64, 1), 128, 0, stream>>>(
            h2a, fcWp, fcb, (float*)d_out, HIDF, HIDF, OUT_D);
}
